// VarianceAdaptor_13451837571680
// MI455X (gfx1250) — matrix operand order repaired and checked by reading, not compiled, not run on hardware
//
#include <hip/hip_runtime.h>
#include <stdint.h>

#define B_ 32
#define S_ 512
#define D_ 512
#define FF 512
#define T_ 2048

typedef __attribute__((ext_vector_type(16))) __bf16 v16bf;
typedef __attribute__((ext_vector_type(8)))  float  v8f;
typedef __attribute__((ext_vector_type(4)))  unsigned int u32x4;

union FragU { u32x4 u[2]; v16bf v; };

__device__ __forceinline__ unsigned short f2bf(float f) {
  unsigned int u = __float_as_uint(f);
  u += 0x7FFFu + ((u >> 16) & 1u);
  return (unsigned short)(u >> 16);
}

// ---------------- fp32 -> bf16 convert ----------------
__global__ __launch_bounds__(256) void cvt_bf16_kernel(const float* __restrict__ in,
                                                       unsigned short* __restrict__ out,
                                                       int n) {
  int i = blockIdx.x * 256 + threadIdx.x;
  if (i < n) out[i] = f2bf(in[i]);
}

// ---------------- weight swizzle into WMMA B-fragment order ----------------
// src: w[f, c, k] (f<FF, c<D_, k<3). dst: chunk = (k*32 + nt)*16 + cc,
// per-chunk 32 lanes x 16 bf16 contiguous. lane: n = lane&15 (f = nt*16+n),
// K-half base kb = (lane>>4)*16. element (v,h): kk = kb + 2v + h, c = cc*32 + kk.
__global__ __launch_bounds__(256) void swz_w_kernel(const float* __restrict__ w,
                                                    unsigned short* __restrict__ dst) {
  int e = blockIdx.x * 256 + threadIdx.x;   // < 3*32*16*32*16 = 786432
  int h    = e & 1;
  int v    = (e >> 1) & 7;
  int lane = (e >> 4) & 31;
  int cc   = (e >> 9) & 15;
  int nt   = (e >> 13) & 31;
  int k    = e >> 18;
  if (k >= 3) return;
  int f  = nt * 16 + (lane & 15);
  int kk = ((lane >> 4) << 4) + 2 * v + h;
  int c  = cc * 32 + kk;
  dst[e] = f2bf(w[((size_t)f * D_ + c) * 3 + k]);
}

// ---------------- conv1d (K=3, pad=1) as WMMA GEMM, bias+ReLU fused ----------------
// One wave computes a 4x(16-row) M group x four 16-col N tiles (16 accumulators).
// B fragment loaded once per (k,cc,j) and reused across the 4 M sub-tiles.
__global__ __launch_bounds__(256) void conv_wmma_kernel(
    const unsigned short* __restrict__ xin,   // [B*S, D_] bf16
    const unsigned short* __restrict__ wswz,  // swizzled bf16 weights
    const float* __restrict__ bias,           // [FF]
    float* __restrict__ out)                  // [B*S, FF]
{
  const int lane  = threadIdx.x & 31;
  const int wid   = threadIdx.x >> 5;
  const int wtile = blockIdx.x * 8 + wid;     // 2048 waves total
  const int ntg   = wtile & 7;                // group of 4 n-tiles
  const int mtg   = wtile >> 3;               // 0..255, group of 4 m-tiles (64 rows)
  const int b     = mtg >> 3;                 // 8 groups of 64 rows per batch
  const int s0    = (mtg & 7) << 6;
  const int m     = lane & 15;
  const int kb    = (lane >> 4) << 4;         // A-layout K-half base (0 or 16)

  v8f acc[4][4];
  #pragma unroll
  for (int u = 0; u < 4; ++u)
    #pragma unroll
    for (int j = 0; j < 4; ++j) acc[u][j] = (v8f)0.0f;

  for (int k = 0; k < 3; ++k) {
    const unsigned short* wk = wswz + (size_t)k * (32 * 16 * 512);
    const unsigned short* arow[4];
    bool ok[4];
    #pragma unroll
    for (int u = 0; u < 4; ++u) {
      const int sp = s0 + u * 16 + m + k - 1;
      ok[u] = (sp >= 0) && (sp < S_);
      arow[u] = xin + ((size_t)b * S_ + (size_t)(ok[u] ? sp : 0)) * D_ + kb;
    }
    for (int cc = 0; cc < 16; ++cc) {
      FragU a[4];
      #pragma unroll
      for (int u = 0; u < 4; ++u) {
        if (ok[u]) {
          const u32x4* ap = (const u32x4*)(arow[u] + cc * 32);
          a[u].u[0] = ap[0];   // K = kb+0..7     (c = cc*32 + kb + j)
          a[u].u[1] = ap[1];   // K = kb+8..15    (c = cc*32 + kb + 8 + j)
        } else {
          a[u].u[0] = (u32x4)0u;
          a[u].u[1] = (u32x4)0u;
        }
      }
      #pragma unroll
      for (int j = 0; j < 4; ++j) {
        const int nt = (ntg << 2) + j;
        const u32x4* bp = (const u32x4*)(wk + ((size_t)(nt * 16 + cc) * 32 + lane) * 16);
        FragU bf;
        bf.u[0] = bp[0];
        bf.u[1] = bp[1];
        #pragma unroll
        for (int u = 0; u < 4; ++u) {
          acc[u][j] = __builtin_amdgcn_wmma_f32_16x16x32_bf16(
              false, a[u].v, false, bf.v, (short)0, acc[u][j], false, false);
        }
      }
    }
  }

  // D layout: element i -> row m = i + 8*(lane>>4), col n = (lane&15)
  const int n0 = (ntg << 6) + m;
  const int mo = (lane >> 4) << 3;
  #pragma unroll
  for (int u = 0; u < 4; ++u) {
    float* orow = out + ((size_t)b * S_ + s0 + u * 16 + mo) * FF + n0;
    #pragma unroll
    for (int j = 0; j < 4; ++j) {
      const float bv = bias[n0 + j * 16];
      #pragma unroll
      for (int i = 0; i < 8; ++i) {
        float vv = acc[u][j][i] + bv;
        orow[(size_t)i * FF + j * 16] = fmaxf(vv, 0.0f);
      }
    }
  }
}

// ---------------- LayerNorm (+ optional bf16 out, + optional fused linear 512->1) ----------------
__global__ __launch_bounds__(256) void ln_fused_kernel(
    const float* __restrict__ in,           // [R, FF]
    const float* __restrict__ g,
    const float* __restrict__ be,
    unsigned short* __restrict__ out_bf,    // nullable
    const float* __restrict__ lw,           // nullable (FF x 1)
    const float* __restrict__ lb,           // nullable (1)
    const unsigned char* __restrict__ mask, // nullable, per-row bool
    float* __restrict__ pred)               // nullable, per-row
{
  const int row  = blockIdx.x;
  const int t    = threadIdx.x;
  const int lane = t & 31;
  const int wid  = t >> 5;
  const float* x = in + (size_t)row * FF;
  float v0 = x[t];
  float v1 = x[t + 256];
  float s = v0 + v1;
  float q = v0 * v0 + v1 * v1;
  #pragma unroll
  for (int o = 16; o > 0; o >>= 1) {
    s += __shfl_xor(s, o, 32);
    q += __shfl_xor(q, o, 32);
  }
  __shared__ float rs[8], rq[8];
  if (lane == 0) { rs[wid] = s; rq[wid] = q; }
  __syncthreads();
  float S = 0.f, Q = 0.f;
  #pragma unroll
  for (int i = 0; i < 8; ++i) { S += rs[i]; Q += rq[i]; }
  const float mu = S * (1.0f / FF);
  float var = Q * (1.0f / FF) - mu * mu;
  var = fmaxf(var, 0.0f);
  const float r = rsqrtf(var + 1e-5f);
  const float h0 = (v0 - mu) * r * g[t] + be[t];
  const float h1 = (v1 - mu) * r * g[t + 256] + be[t + 256];
  if (out_bf) {
    out_bf[(size_t)row * FF + t] = f2bf(h0);
    out_bf[(size_t)row * FF + t + 256] = f2bf(h1);
  }
  if (lw) {
    float d = h0 * lw[t] + h1 * lw[t + 256];
    #pragma unroll
    for (int o = 16; o > 0; o >>= 1) d += __shfl_xor(d, o, 32);
    __syncthreads();
    if (lane == 0) rs[wid] = d;
    __syncthreads();
    if (t == 0) {
      float tot = 0.f;
      #pragma unroll
      for (int i = 0; i < 8; ++i) tot += rs[i];
      float o = tot + lb[0];
      if (mask && mask[row]) o = 0.0f;
      pred[row] = o;
    }
  }
}

// ---------------- pitch/energy K=3 embedding adds -> x2 (bf16) and x3 (f32) ----------------
__global__ __launch_bounds__(256) void embed_add_kernel(
    const float* __restrict__ x, const float* __restrict__ pt, const float* __restrict__ et,
    const float* __restrict__ pew, const float* __restrict__ peb,
    const float* __restrict__ eew, const float* __restrict__ eeb,
    unsigned short* __restrict__ x2bf, float* __restrict__ x3)
{
  size_t e = (size_t)blockIdx.x * 256 + threadIdx.x;  // < B*S*D
  int d = (int)(e & (D_ - 1));
  int s = (int)((e >> 9) & (S_ - 1));
  int b = (int)(e >> 18);
  const float* p = pt + (size_t)b * S_;
  const float* q = et + (size_t)b * S_;
  float pm = (s > 0)      ? p[s - 1] : 0.f;
  float pz = p[s];
  float pp = (s < S_ - 1) ? p[s + 1] : 0.f;
  float em = (s > 0)      ? q[s - 1] : 0.f;
  float ez = q[s];
  float ep = (s < S_ - 1) ? q[s + 1] : 0.f;
  float pe = pew[d * 3 + 0] * pm + pew[d * 3 + 1] * pz + pew[d * 3 + 2] * pp + peb[d];
  float ee = eew[d * 3 + 0] * em + eew[d * 3 + 1] * ez + eew[d * 3 + 2] * ep + eeb[d];
  float x2 = x[e] + pe;
  x2bf[e] = f2bf(x2);
  x3[e] = x2 + ee;
}

// ---------------- duration cumsum + mel_len ----------------
__global__ void cumsum_kernel(const int* __restrict__ dur, int* __restrict__ cum,
                              int* __restrict__ melint, float* __restrict__ mellen_out) {
  int b = threadIdx.x;
  if (b < B_) {
    int c = 0;
    const int* d = dur + (size_t)b * S_;
    int* cm = cum + (size_t)b * S_;
    for (int s = 0; s < S_; ++s) { c += d[s]; cm[s] = c; }
    int ml = c < T_ ? c : T_;
    melint[b] = ml;
    mellen_out[b] = (float)ml;
  }
}

// ---------------- length regulation gather ----------------
__global__ __launch_bounds__(256) void gather_kernel(
    const float* __restrict__ x3, const int* __restrict__ cum,
    const int* __restrict__ melint, float* __restrict__ xmel)
{
  const int bt = blockIdx.x;
  const int b = bt >> 11;          // T_ = 2048
  const int t = bt & (T_ - 1);
  __shared__ int sidx, svalid;
  if (threadIdx.x == 0) {
    const int* c = cum + (size_t)b * S_;
    int lo = 0, hi = S_;
    while (lo < hi) { int mid = (lo + hi) >> 1; if (c[mid] <= t) lo = mid + 1; else hi = mid; }
    sidx = lo < (S_ - 1) ? lo : (S_ - 1);
    svalid = (t < melint[b]) ? 1 : 0;
  }
  __syncthreads();
  const int idx = sidx;
  const int valid = svalid;
  const float* src = x3 + ((size_t)b * S_ + idx) * D_;
  float* dst = xmel + ((size_t)b * T_ + t) * D_;
  #pragma unroll
  for (int r = 0; r < 2; ++r) {
    int d = threadIdx.x + r * 256;
    dst[d] = valid ? src[d] : 0.0f;
  }
}

// ---------------- misc outputs: duration as float, mel_mask zeros ----------------
__global__ __launch_bounds__(256) void misc_kernel(const int* __restrict__ dur,
                                                   float* __restrict__ durf,
                                                   float* __restrict__ melmask) {
  int i = blockIdx.x * 256 + threadIdx.x;   // < B_*T_
  if (i < B_ * S_) durf[i] = (float)dur[i];
  melmask[i] = 0.0f;
}

extern "C" void kernel_launch(void* const* d_in, const int* in_sizes, int n_in,
                              void* d_out, int out_size, void* d_ws, size_t ws_size,
                              hipStream_t stream) {
  (void)in_sizes; (void)n_in; (void)out_size; (void)ws_size;
  const float* x  = (const float*)d_in[0];
  const unsigned char* src_mask = (const unsigned char*)d_in[1];
  const float* pt = (const float*)d_in[4];
  const float* et = (const float*)d_in[5];
  const int* dur  = (const int*)d_in[6];
  const float* pit_ew = (const float*)d_in[37];
  const float* pit_eb = (const float*)d_in[38];
  const float* eng_ew = (const float*)d_in[39];
  const float* eng_eb = (const float*)d_in[40];

  float* out       = (float*)d_out;
  float* o_xmel    = out;
  float* o_pitch   = out + (size_t)B_ * T_ * D_;
  float* o_energy  = o_pitch + B_ * S_;
  float* o_logdur  = o_energy + B_ * S_;
  float* o_durf    = o_logdur + B_ * S_;
  float* o_mellen  = o_durf + B_ * S_;
  float* o_melmask = o_mellen + B_;

  char* ws = (char*)d_ws;
  size_t off = 0;
  auto take = [&](size_t bytes) -> char* {
    char* p = ws + off;
    off = (off + bytes + 255) & ~(size_t)255;
    return p;
  };
  unsigned short* xbf    = (unsigned short*)take((size_t)B_ * S_ * D_ * 2);
  unsigned short* x2bf   = (unsigned short*)take((size_t)B_ * S_ * D_ * 2);
  unsigned short* bfA    = (unsigned short*)take((size_t)B_ * S_ * FF * 2);
  float*          convout= (float*)take((size_t)B_ * S_ * FF * 4);
  float*          x3     = (float*)take((size_t)B_ * S_ * D_ * 4);
  int*            cum    = (int*)take((size_t)B_ * S_ * 4);
  int*            melint = (int*)take(256);
  unsigned short* wz[6];
  for (int i = 0; i < 6; ++i) wz[i] = (unsigned short*)take((size_t)3 * FF * D_ * 2);

  // 1. x -> bf16
  {
    int n = B_ * S_ * D_;
    cvt_bf16_kernel<<<(n + 255) / 256, 256, 0, stream>>>(x, xbf, n);
  }
  // 2. swizzle the 6 conv weights (dur_w1, dur_w2, pit_w1, pit_w2, eng_w1, eng_w2)
  const int wbase[6] = {7, 11, 17, 21, 27, 31};
  for (int i = 0; i < 6; ++i)
    swz_w_kernel<<<786432 / 256, 256, 0, stream>>>((const float*)d_in[wbase[i]], wz[i]);
  // 3. pitch/energy embedding adds -> x2bf (energy-predictor input), x3 (final f32 x)
  embed_add_kernel<<<(B_ * S_ * D_) / 256, 256, 0, stream>>>(
      x, pt, et, pit_ew, pit_eb, eng_ew, eng_eb, x2bf, x3);
  // 4. duration cumsum + mel_len
  cumsum_kernel<<<1, 32, 0, stream>>>(dur, cum, melint, o_mellen);

  // 5. three predictors (dur/pit on original x; energy on x after pitch add)
  struct Pred { int base; const unsigned short* xin; int w1; int w2; float* pred; };
  Pred preds[3] = {
      {7,  xbf,  0, 1, o_logdur},
      {17, xbf,  2, 3, o_pitch},
      {27, x2bf, 4, 5, o_energy},
  };
  for (int p = 0; p < 3; ++p) {
    const int base = preds[p].base;
    const float* b1  = (const float*)d_in[base + 1];
    const float* g1  = (const float*)d_in[base + 2];
    const float* be1 = (const float*)d_in[base + 3];
    const float* b2  = (const float*)d_in[base + 5];
    const float* g2  = (const float*)d_in[base + 6];
    const float* be2 = (const float*)d_in[base + 7];
    const float* lw  = (const float*)d_in[base + 8];
    const float* lb  = (const float*)d_in[base + 9];

    conv_wmma_kernel<<<256, 256, 0, stream>>>(preds[p].xin, wz[preds[p].w1], b1, convout);
    ln_fused_kernel<<<B_ * S_, 256, 0, stream>>>(convout, g1, be1, bfA,
                                                 nullptr, nullptr, nullptr, nullptr);
    conv_wmma_kernel<<<256, 256, 0, stream>>>(bfA, wz[preds[p].w2], b2, convout);
    ln_fused_kernel<<<B_ * S_, 256, 0, stream>>>(convout, g2, be2, nullptr,
                                                 lw, lb, src_mask, preds[p].pred);
  }

  // 6. length regulation gather
  gather_kernel<<<B_ * T_, 256, 0, stream>>>(x3, cum, melint, o_xmel);
  // 7. misc outputs
  misc_kernel<<<(B_ * T_) / 256, 256, 0, stream>>>(dur, o_durf, o_melmask);
}
